// VisionTransformer_59949153518124
// MI455X (gfx1250) — matrix operand order repaired and checked by reading, not compiled, not run on hardware
//
#include <hip/hip_runtime.h>

// ---------------------------------------------------------------------------
// MI455X (gfx1250) implementation.
//   - global-avg-pool commuted through the patchify conv (74 GF -> ~2 GF)
//   - GEMMs: v_wmma_f32_16x16x32_bf16, one wave = 16x64 output (4 N-tiles,
//     shared A operand), M padded to 16 so the K-loop is branch-free
//   - f32 reductions (pool / layernorm / softmax / head)
// ---------------------------------------------------------------------------

typedef __attribute__((ext_vector_type(16))) __bf16 v16bf;
typedef __attribute__((ext_vector_type(8)))  __bf16 v8bf;
typedef __attribute__((ext_vector_type(8)))  float  v8f;

#define B_    8
#define T_    20
#define BT    160      // B*T frames
#define HW    224
#define D_    1536
#define KC    768      // 3*16*16 patch elements
#define NH    12
#define HD    128
#define NC    7
#define NTOK  200      // B*5 groups * 5 tokens
#define NTOKP 208      // padded to 16 rows for branch-free WMMA A loads

static __device__ __forceinline__ __bf16 f2bf(float f) {
  unsigned u = __builtin_bit_cast(unsigned, f);
  u += 0x7FFFu + ((u >> 16) & 1u);              // round to nearest even
  unsigned short s = (unsigned short)(u >> 16);
  return __builtin_bit_cast(__bf16, s);
}

// ---------------- f32 -> bf16 weight conversion ----------------------------
__global__ void k_cvt(const float* __restrict__ in, __bf16* __restrict__ out, int n) {
  int i = blockIdx.x * 256 + threadIdx.x;
  if (i < n) out[i] = f2bf(in[i]);
}

// ---------------- pooled-patch reduction -----------------------------------
// A[f, c*256 + i*16 + j] = mean over 14x14 patches of x[b,c,t,16*pi+i,16*pj+j]
// One block per (frame, c, i); 224 threads cover one image row (w = tid),
// fully coalesced 896B row reads.
__global__ __launch_bounds__(224) void k_pool(const float* __restrict__ x,
                                              __bf16* __restrict__ A) {
  __shared__ float red[224];
  int blk = blockIdx.x;                 // BT * 48
  int bt  = blk / 48;
  int sub = blk % 48;                   // c*16 + i
  int c = sub >> 4, i = sub & 15;
  int b = bt / T_, t = bt % T_;
  int tid = threadIdx.x;                // == w = 16*pj + j
  const float* base = x + ((((size_t)b * 3 + c) * T_ + t) * HW + i) * HW;
  float s = 0.f;
  for (int pi = 0; pi < 14; ++pi)
    s += base[(size_t)pi * 16 * HW + tid];
  red[tid] = s;
  __syncthreads();
  if (tid < 16) {                       // j = tid: sum over the 14 pj groups
    float acc = 0.f;
    for (int p = 0; p < 14; ++p) acc += red[p * 16 + tid];
    A[(size_t)bt * KC + c * 256 + i * 16 + tid] = f2bf(acc * (1.0f / 196.0f));
  }
}

// ---------------- WMMA GEMM: C[M,N] = A[M,K] * W[N,K]^T (+bias)(+resid) -----
// One wave computes a 16x64 strip: A fetched once per K-step, 4 B tiles,
// 4 WMMAs.  A buffer must have >= ceil16(M) valid rows (padded; garbage rows
// only feed C rows >= M which are never stored) -> branch-free K-loop, EXEC
// stays all-ones as WMMA requires.
// Layouts used here (contiguous k, same for A and B): A lane(m=l&15, h=l>>4) K slots {16h+v} -> two 16B
// loads; B lane(n,h) K slots {16h+v} -> one 32B load per tile (W kept
// out-features-major so B[k][n] = W[n][k] is contiguous).
__global__ __launch_bounds__(128) void k_gemm(
    const __bf16* __restrict__ A, const __bf16* __restrict__ W,
    const float* __restrict__ bias, const float* __restrict__ resid,
    float* __restrict__ C, __bf16* __restrict__ Cb,
    int M, int N, int K)
{
  int lane   = threadIdx.x & 31;
  int wave   = threadIdx.x >> 5;
  int stripN = N >> 6;                         // 64-wide strips
  int tilesM = (M + 15) >> 4;
  int strip  = blockIdx.x * 4 + wave;
  if (strip >= tilesM * stripN) return;        // wave-uniform
  int tm = strip / stripN, tn = strip % stripN;
  int half = lane >> 4;
  int mn   = lane & 15;
  const __bf16* ap = A + (size_t)(tm * 16 + mn) * K + half * 16;
  const __bf16* wp = W + (size_t)(tn * 64 + mn) * K + half * 16;
  const size_t ws16 = (size_t)16 * K;          // 16 W-rows
  v8f acc0 = {}, acc1 = {}, acc2 = {}, acc3 = {};
  for (int k0 = 0; k0 < K; k0 += 32) {
    v8bf a0 = *(const v8bf*)(ap + k0);
    v8bf a1 = *(const v8bf*)(ap + k0 + 8);
    v16bf av;
#pragma unroll
    for (int q = 0; q < 8; ++q) { av[q] = a0[q]; av[q + 8] = a1[q]; }
    v16bf b0 = *(const v16bf*)(wp + k0);
    v16bf b1 = *(const v16bf*)(wp + ws16 + k0);
    v16bf b2 = *(const v16bf*)(wp + 2 * ws16 + k0);
    v16bf b3 = *(const v16bf*)(wp + 3 * ws16 + k0);
    acc0 = __builtin_amdgcn_wmma_f32_16x16x32_bf16(false, av, false, b0, (short)0, acc0, false, false);
    acc1 = __builtin_amdgcn_wmma_f32_16x16x32_bf16(false, av, false, b1, (short)0, acc1, false, false);
    acc2 = __builtin_amdgcn_wmma_f32_16x16x32_bf16(false, av, false, b2, (short)0, acc2, false, false);
    acc3 = __builtin_amdgcn_wmma_f32_16x16x32_bf16(false, av, false, b3, (short)0, acc3, false, false);
  }
  v8f accs[4] = {acc0, acc1, acc2, acc3};
#pragma unroll
  for (int j = 0; j < 4; ++j) {
    int col = tn * 64 + j * 16 + mn;
    float bval = bias ? bias[col] : 0.f;
#pragma unroll
    for (int r = 0; r < 8; ++r) {
      int row = tm * 16 + r + 8 * half;        // C/D VGPR layout
      if (row < M) {
        float v = accs[j][r] + bval;
        if (resid) v += resid[(size_t)row * N + col];
        if (C)  C[(size_t)row * N + col]  = v;
        if (Cb) Cb[(size_t)row * N + col] = f2bf(v);
      }
    }
  }
}

// ---------------- group assembly -------------------------------------------
__global__ void k_build(const float* __restrict__ xr, const float* __restrict__ temb,
                        const float* __restrict__ ts, const float* __restrict__ wnum,
                        const float* __restrict__ bnum, const float* __restrict__ cls,
                        float* __restrict__ g) {
  int idx = blockIdx.x * 256 + threadIdx.x;
  if (idx >= B_ * 25 * D_) return;
  int dcol = idx % D_;
  int r = idx / D_;                 // b*25 + n*5 + t'
  int tp = r % 5, n = (r / 5) % 5, b = r / 25;
  float v;
  if (tp == 0) {
    v = cls[(size_t)n * D_ + dcol];
  } else {
    int tok = n * 4 + (tp - 1);
    v = xr[(size_t)(b * T_ + tok) * D_ + dcol] + temb[(size_t)tok * D_ + dcol]
        + ts[b] * wnum[dcol] + bnum[dcol];
  }
  g[idx] = v;
}

// ---------------- layernorm -> bf16 ----------------------------------------
__global__ __launch_bounds__(256) void k_ln(const float* __restrict__ g,
                                            const float* __restrict__ gamma,
                                            const float* __restrict__ beta,
                                            __bf16* __restrict__ out) {
  __shared__ float s1[256], s2[256];
  int row = blockIdx.x;
  const float* xp = g + (size_t)row * D_;
  int tid = threadIdx.x;
  float a = 0.f, b = 0.f;
  for (int i = tid; i < D_; i += 256) { float v = xp[i]; a += v; b += v * v; }
  s1[tid] = a; s2[tid] = b; __syncthreads();
  for (int off = 128; off > 0; off >>= 1) {
    if (tid < off) { s1[tid] += s1[tid + off]; s2[tid] += s2[tid + off]; }
    __syncthreads();
  }
  float mean = s1[0] * (1.0f / D_);
  float var  = s2[0] * (1.0f / D_) - mean * mean;
  float inv  = rsqrtf(var + 1e-5f);
  for (int i = tid; i < D_; i += 256)
    out[(size_t)row * D_ + i] = f2bf((xp[i] - mean) * inv * gamma[i] + beta[i]);
}

// ---------------- tiny attention (seq-len 5, HD=128) ------------------------
__global__ __launch_bounds__(128) void k_attn(const float* __restrict__ qkv,
                                              __bf16* __restrict__ obf) {
  __shared__ float qs[5][HD], ks[5][HD], vs[5][HD], at[5][5];
  int bq = blockIdx.x / NH;
  int h  = blockIdx.x % NH;
  int d  = threadIdx.x;
  for (int t = 0; t < 5; ++t) {
    const float* rp = qkv + (size_t)(bq * 5 + t) * (3 * D_) + h * HD + d;
    qs[t][d] = rp[0]; ks[t][d] = rp[D_]; vs[t][d] = rp[2 * D_];
  }
  __syncthreads();
  if (d < 25) {
    int t = d / 5, u = d % 5;
    float s = 0.f;
    for (int i = 0; i < HD; ++i) s += qs[t][i] * ks[u][i];
    at[t][u] = s * 0.08838834764831845f;          // 128^-0.5
  }
  __syncthreads();
  if (d < 5) {
    float mx = at[d][0];
    for (int u = 1; u < 5; ++u) mx = fmaxf(mx, at[d][u]);
    float sum = 0.f;
    for (int u = 0; u < 5; ++u) { float e = __expf(at[d][u] - mx); at[d][u] = e; sum += e; }
    float r = 1.0f / sum;
    for (int u = 0; u < 5; ++u) at[d][u] *= r;
  }
  __syncthreads();
  for (int t = 0; t < 5; ++t) {
    float o = 0.f;
    for (int u = 0; u < 5; ++u) o += at[t][u] * vs[u][d];
    obf[(size_t)(bq * 5 + t) * D_ + h * HD + d] = f2bf(o);
  }
}

// ---------------- cls-token roll (two-phase to avoid races) -----------------
__global__ void k_cls_save(const float* __restrict__ g, float* __restrict__ tmp) {
  int idx = blockIdx.x * 256 + threadIdx.x;
  if (idx >= B_ * 5 * D_) return;
  int dcol = idx % D_; int r = idx / D_; int n = r % 5, b = r / 5;
  tmp[idx] = g[((size_t)(b * 25 + n * 5)) * D_ + dcol];
}
__global__ void k_cls_roll(const float* __restrict__ tmp, float* __restrict__ g) {
  int idx = blockIdx.x * 256 + threadIdx.x;
  if (idx >= B_ * 5 * D_) return;
  int dcol = idx % D_; int r = idx / D_; int n = r % 5, b = r / 5;
  int src = (n + 4) % 5;                         // roll(+1)
  g[((size_t)(b * 25 + n * 5)) * D_ + dcol] = tmp[((size_t)(b * 5 + src)) * D_ + dcol];
}

// ---------------- classifier head ------------------------------------------
__global__ __launch_bounds__(256) void k_head(const float* __restrict__ g,
                                              const float* __restrict__ xr,
                                              const float* __restrict__ wh,
                                              const float* __restrict__ bh,
                                              float* __restrict__ out) {
  __shared__ float s[256];
  int b = blockIdx.x / NC, c = blockIdx.x % NC;
  int tid = threadIdx.x;
  const float* gp = g  + (size_t)(b * 25 + 24) * D_;   // n=4, t'=4 == token 19
  const float* xp = xr + (size_t)(b * T_ + 19) * D_;
  const float* wp = wh + (size_t)c * D_;
  float a = 0.f;
  for (int i = tid; i < D_; i += 256) a += (gp[i] + xp[i]) * wp[i];
  s[tid] = a; __syncthreads();
  for (int off = 128; off > 0; off >>= 1) {
    if (tid < off) s[tid] += s[tid + off];
    __syncthreads();
  }
  if (tid == 0) out[b * NC + c] = s[0] + bh[c];
}

// ---------------------------------------------------------------------------
extern "C" void kernel_launch(void* const* d_in, const int* in_sizes, int n_in,
                              void* d_out, int out_size, void* d_ws, size_t ws_size,
                              hipStream_t stream) {
  const float* x      = (const float*)d_in[0];
  const float* ts     = (const float*)d_in[1];
  const float* conv_w = (const float*)d_in[2];
  const float* conv_b = (const float*)d_in[3];
  const float* w_red  = (const float*)d_in[4];
  const float* b_red  = (const float*)d_in[5];
  const float* w_num  = (const float*)d_in[6];
  const float* b_num  = (const float*)d_in[7];
  const float* temb   = (const float*)d_in[8];
  const float* cls    = (const float*)d_in[9];
  const float* ln_g   = (const float*)d_in[10];
  const float* ln_b   = (const float*)d_in[11];
  const float* w_qkv  = (const float*)d_in[12];
  const float* w_tpr  = (const float*)d_in[13];
  const float* b_tpr  = (const float*)d_in[14];
  const float* w_tfc  = (const float*)d_in[15];
  const float* b_tfc  = (const float*)d_in[16];
  const float* w_head = (const float*)d_in[17];
  const float* b_head = (const float*)d_in[18];
  float* out = (float*)d_out;

  char* ws = (char*)d_ws;
  size_t off = 0;
  auto alloc = [&](size_t bytes) -> char* {
    char* p = ws + off; off = (off + bytes + 255) & ~(size_t)255; return p;
  };
  __bf16* A_bf    = (__bf16*)alloc((size_t)BT * KC * 2);
  __bf16* cw_bf   = (__bf16*)alloc((size_t)D_ * KC * 2);
  __bf16* wred_bf = (__bf16*)alloc((size_t)D_ * D_ * 2);
  __bf16* wqkv_bf = (__bf16*)alloc((size_t)3 * D_ * D_ * 2);
  __bf16* wtpr_bf = (__bf16*)alloc((size_t)D_ * D_ * 2);
  __bf16* wtfc_bf = (__bf16*)alloc((size_t)D_ * D_ * 2);
  __bf16* f_bf    = (__bf16*)alloc((size_t)BT * D_ * 2);
  float*  xr      = (float*) alloc((size_t)BT * D_ * 4);
  float*  g       = (float*) alloc((size_t)NTOK * D_ * 4);
  __bf16* hn_bf   = (__bf16*)alloc((size_t)NTOKP * D_ * 2);   // padded A operand
  float*  qkv     = (float*) alloc((size_t)NTOK * 3 * D_ * 4);
  __bf16* o_bf    = (__bf16*)alloc((size_t)NTOKP * D_ * 2);   // padded A operand
  __bf16* p_bf    = (__bf16*)alloc((size_t)NTOKP * D_ * 2);   // padded A operand
  float*  ctmp    = (float*) alloc((size_t)B_ * 5 * D_ * 4);

  auto cvt = [&](const float* s, __bf16* d, size_t n) {
    k_cvt<<<dim3((unsigned)((n + 255) / 256)), dim3(256), 0, stream>>>(s, d, (int)n);
  };
  cvt(conv_w, cw_bf,  (size_t)D_ * KC);
  cvt(w_red,  wred_bf,(size_t)D_ * D_);
  cvt(w_qkv,  wqkv_bf,(size_t)3 * D_ * D_);
  cvt(w_tpr,  wtpr_bf,(size_t)D_ * D_);
  cvt(w_tfc,  wtfc_bf,(size_t)D_ * D_);

  k_pool<<<dim3(BT * 48), dim3(224), 0, stream>>>(x, A_bf);

  auto gemm = [&](const __bf16* A, const __bf16* W, const float* bias,
                  const float* resid, float* C, __bf16* Cb, int M, int N, int K) {
    int strips = (((M + 15) / 16)) * (N / 64);
    k_gemm<<<dim3((strips + 3) / 4), dim3(128), 0, stream>>>(A, W, bias, resid, C, Cb, M, N, K);
  };

  // pooled-conv + bias  -> f (bf16 only; feeds next GEMM)
  gemm(A_bf, cw_bf, conv_b, nullptr, nullptr, f_bf, BT, D_, KC);
  // reduce_dim -> xr (f32, needed for groups + final residual)
  gemm(f_bf, wred_bf, b_red, nullptr, xr, nullptr, BT, D_, D_);

  { int n = B_ * 25 * D_;
    k_build<<<dim3((n + 255) / 256), dim3(256), 0, stream>>>(xr, temb, ts, w_num, b_num, cls, g); }

  for (int it = 0; it < 4; ++it) {
    k_ln<<<dim3(NTOK), dim3(256), 0, stream>>>(g, ln_g, ln_b, hn_bf);
    gemm(hn_bf, wqkv_bf, nullptr, nullptr, qkv, nullptr, NTOK, 3 * D_, D_);
    k_attn<<<dim3(40 * NH), dim3(HD), 0, stream>>>(qkv, o_bf);
    gemm(o_bf, wtpr_bf, b_tpr, nullptr, nullptr, p_bf, NTOK, D_, D_);
    gemm(p_bf, wtfc_bf, b_tfc, g, g, nullptr, NTOK, D_, D_);   // fc + residual, in place
    int n = B_ * 5 * D_;
    k_cls_save<<<dim3((n + 255) / 256), dim3(256), 0, stream>>>(g, ctmp);
    k_cls_roll<<<dim3((n + 255) / 256), dim3(256), 0, stream>>>(ctmp, g);
  }

  k_head<<<dim3(B_ * NC), dim3(256), 0, stream>>>(g, xr, w_head, b_head, out);

  (void)in_sizes; (void)n_in; (void)out_size; (void)ws_size;
}
